// ModelK_46952582480550
// MI455X (gfx1250) — matrix operand order repaired and checked by reading, not compiled, not run on hardware
//
#include <hip/hip_runtime.h>
#include <hip/hip_bf16.h>
#include <math.h>

#define NNODES 50000
#define NEDGES 400000
#define BN_EPS 1e-5f

typedef __attribute__((ext_vector_type(16))) __bf16 v16bf;
typedef __attribute__((ext_vector_type(8)))  float  v8f;

__device__ __forceinline__ unsigned short f2bf(float f) {
  unsigned u = __float_as_uint(f);
  u += 0x7fffu + ((u >> 16) & 1u);       // round-to-nearest-even
  return (unsigned short)(u >> 16);
}

union Frag {
  v16bf v;
  unsigned short s[16];
  uint4 q[2];
};

// ---------------------------------------------------------------------------
// Pack a [K,N] f32 row-major weight into bf16 WMMA B-fragment layout:
// out[((nt*KT + kt)*512) + lane*16 + i] = W[kt*32 + (lane>=16?8:0) + (i&7) + (i>=8?16:0)][nt*16 + (lane&15)]
// ---------------------------------------------------------------------------
__global__ void k_pack(const float* __restrict__ W, unsigned short* __restrict__ out,
                       int K, int N, int KT) {
  int t = blockIdx.x * 256 + threadIdx.x;
  int total = KT * (N >> 4) * 512;
  if (t >= total) return;
  int i = t & 15;
  int lane = (t >> 4) & 31;
  int frag = t >> 9;
  int kt = frag % KT;
  int nt = frag / KT;
  int n = nt * 16 + (lane & 15);
  int k = kt * 32 + ((lane & 16) ? 8 : 0) + (i & 7) + ((i & 8) ? 16 : 0);
  float v = (k < K) ? W[(size_t)k * N + n] : 0.f;
  out[t] = f2bf(v);
}

// ---------------------------------------------------------------------------
// bf16 WMMA GEMM: C[M, 128*gridDim.y] = A[M,K] (f32 -> bf16) @ Bpacked
// 256 threads = 8 waves; wave w owns rows [blk*128 + w*16, +16), 8 col tiles.
// GATHER_A: A rows are [h[src[r]] || h[dst[r]]], K=256, GA has 128 cols.
// GADD:     epilogue += addB[dst[row]*512 + c] + addC[src[row]*512 + c]
// ---------------------------------------------------------------------------
template <bool GATHER_A, bool GADD, bool RELU>
__global__ __launch_bounds__(256, 2) void gemm_bf16_wmma(
    const float* __restrict__ A, int lda,
    const float* __restrict__ GA,
    const int* __restrict__ gsrc, const int* __restrict__ gdst,
    const unsigned short* __restrict__ Bp,
    int M, int K, int KT,
    const float* __restrict__ bias,
    const float* __restrict__ addB, const float* __restrict__ addC,
    float* __restrict__ Cout, int ldc) {
  const int lane = threadIdx.x & 31;
  const int wave = threadIdx.x >> 5;
  const int m0 = blockIdx.x * 128 + wave * 16;
  const int l15 = lane & 15;
  const int hi = (lane >> 4) & 1;
  const int koff = hi ? 8 : 0;

  int arow = m0 + l15;
  if (arow > M - 1) arow = M - 1;

  const float* abase = nullptr;
  const float* sbase = nullptr;
  const float* dbase = nullptr;
  if (GATHER_A) {
    sbase = GA + (size_t)gsrc[arow] * 128;
    dbase = GA + (size_t)gdst[arow] * 128;
  } else {
    abase = A + (size_t)arow * lda;
  }

  v8f acc[8];
#pragma unroll
  for (int i = 0; i < 8; ++i) acc[i] = (v8f)0.f;

  const int ntbase = blockIdx.y * 8;

  for (int kt = 0; kt < KT; ++kt) {
    Frag a;
    const int kb = kt * 32;
    if (GATHER_A) {
      const float* p = (kb < 128) ? (sbase + kb) : (dbase + kb - 128);
#pragma unroll
      for (int j = 0; j < 8; ++j) {
        a.s[j]     = f2bf(p[koff + j]);
        a.s[8 + j] = f2bf(p[16 + koff + j]);
      }
    } else if (kb + 32 <= K) {
      const float* p = abase + kb;
#pragma unroll
      for (int j = 0; j < 8; ++j) {
        a.s[j]     = f2bf(p[koff + j]);
        a.s[8 + j] = f2bf(p[16 + koff + j]);
      }
    } else {
#pragma unroll
      for (int j = 0; j < 8; ++j) {
        int k0 = kb + koff + j;
        int k1 = kb + 16 + koff + j;
        a.s[j]     = (k0 < K) ? f2bf(abase[k0]) : (unsigned short)0;
        a.s[8 + j] = (k1 < K) ? f2bf(abase[k1]) : (unsigned short)0;
      }
    }

#pragma unroll
    for (int nt = 0; nt < 8; ++nt) {
      Frag b;
      const uint4* bp = reinterpret_cast<const uint4*>(
          Bp + ((size_t)((ntbase + nt) * KT + kt) * 512 + lane * 16));
      b.q[0] = bp[0];
      b.q[1] = bp[1];
      acc[nt] = __builtin_amdgcn_wmma_f32_16x16x32_bf16(
          false, a.v, false, b.v, (short)0, acc[nt], false, false);
    }
  }

#pragma unroll
  for (int r = 0; r < 8; ++r) {
    const int row = m0 + r + hi * 8;
    if (row >= M) continue;
    const float* pB = nullptr;
    const float* pC = nullptr;
    if (GADD) {
      pB = addB + (size_t)gdst[row] * 512;
      pC = addC + (size_t)gsrc[row] * 512;
    }
#pragma unroll
    for (int nt = 0; nt < 8; ++nt) {
      const int c = l15 + nt * 16;
      const int col = blockIdx.y * 128 + c;
      float v = acc[nt][r];
      if (GADD) v += pB[c] + pC[c];
      if (bias) v += bias[col];
      if (RELU) v = fmaxf(v, 0.f);
      Cout[(size_t)row * ldc + col] = v;
    }
  }
}

// ---------------------------------------------------------------------------
__global__ void k_fill(float* __restrict__ p, float v, long long n) {
  long long i = (long long)blockIdx.x * 256 + threadIdx.x;
  if (i < n) p[i] = v;
}

__global__ void k_zero256(float* p) { p[threadIdx.x] = 0.f; }

// msg = sigmoid(e) * hV[src]; agg[dst] = max(agg[dst], msg)
__global__ void k_msg_aggmax(const float* __restrict__ e, const float* __restrict__ hV,
                             const int* __restrict__ src, const int* __restrict__ dst,
                             float* __restrict__ agg, long long total) {
  long long i = (long long)blockIdx.x * 256 + threadIdx.x;
  if (i >= total) return;
  int r = (int)(i >> 7), c = (int)(i & 127);
  float w = 1.f / (1.f + __expf(-e[i]));
  float m = hV[(size_t)src[r] * 512 + c] * w;
  __hip_atomic_fetch_max(&agg[(size_t)dst[r] * 128 + c], m,
                         __ATOMIC_RELAXED, __HIP_MEMORY_SCOPE_AGENT);
}

// x = hU + (agg finite ? agg : 0)   (hU = hX col slice 0..127, stride 512)
__global__ void k_make_x(const float* __restrict__ hX, const float* __restrict__ agg,
                         float* __restrict__ x, long long total) {
  long long i = (long long)blockIdx.x * 256 + threadIdx.x;
  if (i >= total) return;
  int r = (int)(i >> 7), c = (int)(i & 127);
  float a = agg[i];
  if (!(a > -1e37f)) a = 0.f;   // -inf (empty segment) -> 0
  x[i] = hX[(size_t)r * 512 + c] + a;
}

// per-channel sum / sumsq over rows; s12[0..127]=sum, s12[128..255]=sumsq
__global__ void k_colstats(const float* __restrict__ X, int Mrows,
                           float* __restrict__ s12) {
  __shared__ float sh[256];
  int c = threadIdx.x & 127, g = threadIdx.x >> 7;
  int row0 = blockIdx.x * 128;
  float a = 0.f, b = 0.f;
  for (int i = g; i < 128; i += 2) {
    int r = row0 + i;
    if (r < Mrows) {
      float v = X[(size_t)r * 128 + c];
      a += v;
      b += v * v;
    }
  }
  sh[threadIdx.x] = a;
  __syncthreads();
  float a2 = 0.f;
  if (g == 0) a2 = sh[threadIdx.x] + sh[threadIdx.x + 128];
  __syncthreads();
  sh[threadIdx.x] = b;
  __syncthreads();
  if (g == 0) {
    float b2 = sh[threadIdx.x] + sh[threadIdx.x + 128];
    atomicAdd(&s12[c], a2);
    atomicAdd(&s12[128 + c], b2);
  }
}

// base += relu((x - mean) * rsqrt(var + eps))
__global__ void k_bn_apply(float* __restrict__ base, const float* __restrict__ x,
                           const float* __restrict__ s12, float invM, long long total) {
  long long i = (long long)blockIdx.x * 256 + threadIdx.x;
  if (i >= total) return;
  int c = (int)(i & 127);
  float m = s12[c] * invM;
  float v = s12[128 + c] * invM - m * m;
  float t = (x[i] - m) * rsqrtf(v + BN_EPS);
  base[i] += fmaxf(t, 0.f);
}

// bias2[j] = W0_b[j] + sum_d moy[d] * W0_w[d,j]   (moy = colsum(h)/N)
__global__ void k_bias2(const float* __restrict__ s12, float invN,
                        const float* __restrict__ W0w, const float* __restrict__ W0b,
                        float* __restrict__ bias2) {
  int j = threadIdx.x;
  float acc = W0b[j];
  for (int d = 0; d < 128; ++d) acc += s12[d] * invN * W0w[d * 128 + j];
  bias2[j] = acc;
}

// out[r] = sigmoid(dot(act[r,:], Wf) + Wfb) — one wave32 per row
__global__ void k_final(const float* __restrict__ act, const float* __restrict__ wf,
                        const float* __restrict__ wfb, float* __restrict__ out, int E) {
  int wave = threadIdx.x >> 5, lane = threadIdx.x & 31;
  int r = blockIdx.x * 8 + wave;
  if (r >= E) return;
  const float* p = act + (size_t)r * 128;
  float acc = 0.f;
#pragma unroll
  for (int j = 0; j < 4; ++j) acc += p[lane + j * 32] * wf[lane + j * 32];
  for (int o = 16; o > 0; o >>= 1) acc += __shfl_xor(acc, o, 32);
  if (lane == 0) out[r] = 1.f / (1.f + __expf(-(acc + wfb[0])));
}

// ---------------------------------------------------------------------------
extern "C" void kernel_launch(void* const* d_in, const int* in_sizes, int n_in,
                              void* d_out, int out_size, void* d_ws, size_t ws_size,
                              hipStream_t stream) {
  (void)in_sizes; (void)n_in; (void)out_size; (void)ws_size;
  const float* in_h    = (const float*)d_in[0];
  const float* in_e    = (const float*)d_in[1];
  const float* emb_n_w = (const float*)d_in[2];
  const float* emb_n_b = (const float*)d_in[3];
  const float* emb_e_w = (const float*)d_in[4];
  const float* emb_e_b = (const float*)d_in[5];
  const float* Uw  = (const float*)d_in[6];
  const float* Vw  = (const float*)d_in[7];
  const float* Aw  = (const float*)d_in[8];
  const float* Bw  = (const float*)d_in[9];
  const float* Cw  = (const float*)d_in[10];
  const float* W0w = (const float*)d_in[11];
  const float* W0b = (const float*)d_in[12];
  const float* Wkw = (const float*)d_in[13];
  const float* Wkb = (const float*)d_in[14];
  const float* Wfw = (const float*)d_in[15];
  const float* Wfb = (const float*)d_in[16];
  const int*   src = (const int*)d_in[17];
  const int*   dst = (const int*)d_in[18];
  float* out = (float*)d_out;

  const int N = NNODES, E = NEDGES;
  char* p = (char*)d_ws;
  auto alloc = [&](size_t bytes) -> char* {
    char* r = p;
    p += (bytes + 255) & ~(size_t)255;
    return r;
  };
  float* h     = (float*)alloc((size_t)N * 128 * 4);
  float* e     = (float*)alloc((size_t)E * 128 * 4);
  float* hX    = (float*)alloc((size_t)N * 512 * 4);   // [U|V|B|C] products
  float* x     = (float*)alloc((size_t)N * 128 * 4);   // BN pre-activation (h side)
  float* agg   = (float*)alloc((size_t)N * 128 * 4);
  float* y     = (float*)alloc((size_t)E * 128 * 4);   // BN pre-activation (e side)
  float* s12   = (float*)alloc(256 * 4);
  float* bias2 = (float*)alloc(128 * 4);
  unsigned short* embnP   = (unsigned short*)alloc(8 * 512 * 2);
  unsigned short* embeP   = (unsigned short*)alloc(8 * 512 * 2);
  unsigned short* WcatP0  = (unsigned short*)alloc(65536 * 2);
  unsigned short* WcatP1  = (unsigned short*)alloc(65536 * 2);
  unsigned short* ApP0    = (unsigned short*)alloc(16384 * 2);
  unsigned short* ApP1    = (unsigned short*)alloc(16384 * 2);
  unsigned short* W0P     = (unsigned short*)alloc(32768 * 2);
  unsigned short* Wk0P    = (unsigned short*)alloc(16384 * 2);
  unsigned short* Wk1P    = (unsigned short*)alloc(16384 * 2);
  unsigned short* WcatP[2] = {WcatP0, WcatP1};
  unsigned short* ApP[2]   = {ApP0, ApP1};

  auto pack = [&](const float* W, unsigned short* o, int K, int Ncols, int KT) {
    int total = KT * (Ncols >> 4) * 512;
    k_pack<<<(total + 255) / 256, 256, 0, stream>>>(W, o, K, Ncols, KT);
  };

  // ---- weight packing (bf16 WMMA fragment layout) ----
  pack(emb_n_w, embnP, 16, 128, 1);
  pack(emb_e_w, embeP, 8, 128, 1);
  for (int l = 0; l < 2; ++l) {
    pack(Uw + l * 16384, WcatP[l] +     0, 128, 128, 4);  // cols 0..127
    pack(Vw + l * 16384, WcatP[l] + 16384, 128, 128, 4);  // cols 128..255
    pack(Bw + l * 16384, WcatP[l] + 32768, 128, 128, 4);  // cols 256..383
    pack(Cw + l * 16384, WcatP[l] + 49152, 128, 128, 4);  // cols 384..511
    pack(Aw + l * 16384, ApP[l], 128, 128, 4);
  }
  pack(W0w + 128 * 128, W0P, 256, 128, 8);   // rows 128..383 (hi||hj part)
  pack(Wkw,          Wk0P, 128, 128, 4);
  pack(Wkw + 16384,  Wk1P, 128, 128, 4);

  const int mbN = (N + 127) / 128;            // 391
  const int mbE = E / 128;                    // 3125
  const long long totN = (long long)N * 128;  // 6.4M
  const long long totE = (long long)E * 128;  // 51.2M
  const int gN = (int)((totN + 255) / 256);
  const int gE = (int)((totE + 255) / 256);

  // ---- input embeddings ----
  gemm_bf16_wmma<false, false, false><<<dim3(mbN, 1), 256, 0, stream>>>(
      in_h, 16, nullptr, nullptr, nullptr, embnP, N, 16, 1, emb_n_b,
      nullptr, nullptr, h, 128);
  gemm_bf16_wmma<false, false, false><<<dim3(mbE, 1), 256, 0, stream>>>(
      in_e, 8, nullptr, nullptr, nullptr, embeP, E, 8, 1, emb_e_b,
      nullptr, nullptr, e, 128);

  // ---- message-passing layers ----
  for (int l = 0; l < 2; ++l) {
    // hX = h @ [U|V|B|C]  (N x 512)
    gemm_bf16_wmma<false, false, false><<<dim3(mbN, 4), 256, 0, stream>>>(
        h, 128, nullptr, nullptr, nullptr, WcatP[l], N, 128, 4, nullptr,
        nullptr, nullptr, hX, 512);
    // agg = segment_max(sigmoid(e) * hV[src], dst)
    k_fill<<<gN, 256, 0, stream>>>(agg, -__builtin_inff(), totN);
    k_msg_aggmax<<<gE, 256, 0, stream>>>(e, hX + 128, src, dst, agg, totE);
    // x = hU + fixup(agg); BN stats; h += relu(bn(x))
    k_make_x<<<gN, 256, 0, stream>>>(hX, agg, x, totN);
    k_zero256<<<1, 256, 0, stream>>>(s12);
    k_colstats<<<mbN, 256, 0, stream>>>(x, N, s12);
    k_bn_apply<<<gN, 256, 0, stream>>>(h, x, s12, 1.f / N, totN);
    // y = e @ A + hB[dst] + hC[src]; BN stats; e += relu(bn(y))
    gemm_bf16_wmma<false, true, false><<<dim3(mbE, 1), 256, 0, stream>>>(
        e, 128, nullptr, src, dst, ApP[l], E, 128, 4, nullptr,
        hX + 256, hX + 384, y, 128);
    k_zero256<<<1, 256, 0, stream>>>(s12);
    k_colstats<<<mbE, 256, 0, stream>>>(y, E, s12);
    k_bn_apply<<<gE, 256, 0, stream>>>(e, y, s12, 1.f / E, totE);
  }

  // ---- readout MLP ----
  k_zero256<<<1, 256, 0, stream>>>(s12);
  k_colstats<<<mbN, 256, 0, stream>>>(h, N, s12);          // colsum(h) for moy
  k_bias2<<<1, 128, 0, stream>>>(s12, 1.f / N, W0w, W0b, bias2);

  float* bufA = y;  // e-side BN scratch is dead now
  float* bufB = e;  // e is dead after layers
  // out0 = relu([h[src] || h[dst]] @ W0[128:,:] + (W0_b + moy @ W0[:128,:]))
  gemm_bf16_wmma<true, false, true><<<dim3(mbE, 1), 256, 0, stream>>>(
      nullptr, 0, h, src, dst, W0P, E, 256, 8, bias2,
      nullptr, nullptr, bufA, 128);
  gemm_bf16_wmma<false, false, true><<<dim3(mbE, 1), 256, 0, stream>>>(
      bufA, 128, nullptr, nullptr, nullptr, Wk0P, E, 128, 4, Wkb,
      nullptr, nullptr, bufB, 128);
  gemm_bf16_wmma<false, false, true><<<dim3(mbE, 1), 256, 0, stream>>>(
      bufB, 128, nullptr, nullptr, nullptr, Wk1P, E, 128, 4, Wkb + 128,
      nullptr, nullptr, bufA, 128);
  k_final<<<E / 8, 256, 0, stream>>>(bufA, Wfw, Wfb, out, E);
}
